// Attention_49400713838765
// MI455X (gfx1250) — compile-verified, matrix operand order checked by reading, not run on hardware
//
#include <hip/hip_runtime.h>
#include <hip/hip_bf16.h>

// ---------------------------------------------------------------------------
// CDNA5 (gfx1250) fused attention pipeline, bf16 WMMA with fp32 accumulate.
//   B=1, N=2048, D=2048, H=16, head_dim=128
// ---------------------------------------------------------------------------

typedef __attribute__((ext_vector_type(16))) __bf16 v16bf;
typedef __attribute__((ext_vector_type(8)))  float  v8f;

union Frag {              // one 16x32 bf16 WMMA operand: 32 bytes / lane
    uint4  u[2];
    v16bf  v;
};

__device__ __forceinline__ v8f wmma_bf16(const Frag& a, const Frag& b, v8f c) {
    // 8 args: (neg_a, A, neg_b, B, c_mod, C, reuse_a, reuse_b)
    return __builtin_amdgcn_wmma_f32_16x16x32_bf16(false, a.v, false, b.v,
                                                   (short)0, c, false, false);
}

// ---------------------------------------------------------------------------
// fp32 -> bf16 conversion (vectorized, n must be a multiple of 4)
// ---------------------------------------------------------------------------
__global__ void cvt_f32_bf16(const float* __restrict__ in,
                             __bf16* __restrict__ out, int n4) {
    int i = blockIdx.x * blockDim.x + threadIdx.x;
    if (i < n4) {
        float4 f = ((const float4*)in)[i];
        union { __bf16 h[4]; uint2 u; } p;
        p.h[0] = (__bf16)f.x; p.h[1] = (__bf16)f.y;
        p.h[2] = (__bf16)f.z; p.h[3] = (__bf16)f.w;
        *(uint2*)(out + (size_t)i * 4) = p.u;
    }
}

// ---------------------------------------------------------------------------
// Tiled bf16 GEMM: C[M,N] = A[M,K] * B[K,N], fp32 accumulate.
// Block = 256 threads = 8 waves; macro tile 64x128, K step 32.
// Wave (wm,wn): wm in {0,1} over 32-row halves, wn in 0..3 over 32-col slabs.
// Each wave: 2 A-frags x 2 B-frags -> 4 WMMAs per k-step (4 16x16 C tiles).
// ---------------------------------------------------------------------------
template <bool OUT_BF16>
__global__ __launch_bounds__(256, 1)
void gemm_bf16_wmma(const __bf16* __restrict__ A,
                    const __bf16* __restrict__ B,
                    void* __restrict__ C,
                    int M, int N, int Kd) {
    constexpr int BM = 64, BN = 128, BK = 32;
    __shared__ __align__(16) __bf16 As[BM][BK];   // row-major, reduction contiguous
    __shared__ __align__(16) __bf16 Bs[BN][BK];   // transposed: [n][k], reduction contiguous

    const int tid  = threadIdx.x;
    const int lane = tid & 31;
    const int wave = tid >> 5;
    const int g    = lane >> 4;        // half-wave (ISA lane-group)
    const int l16  = lane & 15;
    const int mBlk = blockIdx.y * BM;
    const int nBlk = blockIdx.x * BN;
    const int wm   = wave & 1;         // 2 wave-rows  (32 rows each)
    const int wn   = wave >> 1;        // 4 wave-cols  (32 cols each)

    v8f acc00 = {}, acc01 = {}, acc10 = {}, acc11 = {};

    const int arow = tid >> 2, aseg = (tid & 3) * 8;    // 64 rows x 4 segs of 8
    const int brow = tid >> 3, bseg = (tid & 7) * 16;   // 32 rows x 8 segs of 16

    for (int k0 = 0; k0 < Kd; k0 += BK) {
        // stage A tile (vector 16B copies)
        *(uint4*)&As[arow][aseg] =
            *(const uint4*)&A[(size_t)(mBlk + arow) * Kd + k0 + aseg];
        // stage B tile transposed into [n][k]
        union { uint4 u[2]; __bf16 h[16]; } bl;
        bl.u[0] = *(const uint4*)&B[(size_t)(k0 + brow) * N + nBlk + bseg];
        bl.u[1] = *(const uint4*)&B[(size_t)(k0 + brow) * N + nBlk + bseg + 8];
#pragma unroll
        for (int j = 0; j < 16; ++j) Bs[bseg + j][brow] = bl.h[j];

        // prefetch next K tile (global_prefetch_b8)
        if (k0 + BK < Kd) {
            __builtin_prefetch(&A[(size_t)(mBlk + arow) * Kd + k0 + BK + aseg], 0, 1);
            __builtin_prefetch(&B[(size_t)(k0 + BK + brow) * N + nBlk + bseg], 0, 1);
        }
        __syncthreads();

        // ISA 16-bit A-layout: lane holds row m = base + l16,
        // reduction indices {8g..8g+7} and {16+8g..16+8g+7} (two b128 LDS reads)
        Frag a0, a1, b0, b1;
        a0.u[0] = *(const uint4*)&As[wm * 32 + l16][g * 8];
        a0.u[1] = *(const uint4*)&As[wm * 32 + l16][16 + g * 8];
        a1.u[0] = *(const uint4*)&As[wm * 32 + 16 + l16][g * 8];
        a1.u[1] = *(const uint4*)&As[wm * 32 + 16 + l16][16 + g * 8];
        b0.u[0] = *(const uint4*)&Bs[wn * 32 + l16][g * 8];
        b0.u[1] = *(const uint4*)&Bs[wn * 32 + l16][16 + g * 8];
        b1.u[0] = *(const uint4*)&Bs[wn * 32 + 16 + l16][g * 8];
        b1.u[1] = *(const uint4*)&Bs[wn * 32 + 16 + l16][16 + g * 8];

        acc00 = wmma_bf16(a0, b0, acc00);
        acc01 = wmma_bf16(a0, b1, acc01);
        acc10 = wmma_bf16(a1, b0, acc10);
        acc11 = wmma_bf16(a1, b1, acc11);
        __syncthreads();
    }

    // C/D layout: VGPR r -> row r + 8g, col = l16
    const int col0 = nBlk + wn * 32 + l16;
    const int col1 = col0 + 16;
#pragma unroll
    for (int r = 0; r < 8; ++r) {
        const int row0 = mBlk + wm * 32 + r + 8 * g;
        const int row1 = row0 + 16;
        if constexpr (OUT_BF16) {
            __bf16* Cb = (__bf16*)C;
            Cb[(size_t)row0 * N + col0] = (__bf16)acc00[r];
            Cb[(size_t)row0 * N + col1] = (__bf16)acc01[r];
            Cb[(size_t)row1 * N + col0] = (__bf16)acc10[r];
            Cb[(size_t)row1 * N + col1] = (__bf16)acc11[r];
        } else {
            float* Cf = (float*)C;
            Cf[(size_t)row0 * N + col0] = acc00[r];
            Cf[(size_t)row0 * N + col1] = acc01[r];
            Cf[(size_t)row1 * N + col0] = acc10[r];
            Cf[(size_t)row1 * N + col1] = acc11[r];
        }
    }
}

// ---------------------------------------------------------------------------
// Causal flash attention, bf16 WMMA.
// qkv: [2048, 6144] bf16 with layout n*6144 + s*2048 + h*128 + d  (s=0:Q,1:K,2:V)
// out: [2048, 2048] bf16 (attn output, head-concatenated)
// Grid: (32 query tiles of 64 rows, 16 heads), block = 128 threads = 4 waves.
// Each wave owns 16 query rows across the full head_dim=128 (8 accumulators).
//
// Key-order permutation trick: the P tile and the V^T tile both index the key
// dimension in interleaved order pkey(k) = (k<16) ? 2k : 2(k-16)+1, so the
// per-lane pair (p0,p1) packs into a single ds_store_b32.  The PV reduction
// is order-invariant, so this is exact.
// ---------------------------------------------------------------------------
__global__ __launch_bounds__(128, 1)
void flash_attn_wmma(const __bf16* __restrict__ qkv,
                     __bf16* __restrict__ outb) {
    constexpr int HD = 128, LDQ = 6144, DMODEL = 2048;
    constexpr float SCALE = 0.08838834764831845f;   // 1/sqrt(128)

    const int h     = blockIdx.y;
    const int qBase = blockIdx.x * 64;
    const int tid   = threadIdx.x;
    const int lane  = tid & 31;
    const int wave  = tid >> 5;
    const int g     = lane >> 4;
    const int l16   = lane & 15;

    __shared__ __align__(16) __bf16 Ks[32][128];      // [key][d]   (reduction=d contiguous)
    __shared__ __align__(16) __bf16 Vt[128][32];      // [d][pkey]  (reduction=key contiguous)
    __shared__ __align__(16) __bf16 Ps[4][16][32];    // per-wave P tile, permuted key order

    // ---- load Q A-fragments once (loop invariant): 4 chunks over d ----
    const int qRowBase = qBase + wave * 16;
    Frag qf[4];
#pragma unroll
    for (int kc = 0; kc < 4; ++kc) {
        const __bf16* src = qkv + (size_t)(qRowBase + l16) * LDQ + h * HD + kc * 32;
        qf[kc].u[0] = *(const uint4*)(src + g * 8);
        qf[kc].u[1] = *(const uint4*)(src + 16 + g * 8);
    }

    v8f oacc[8] = {};
    float mrow[8], lrow[8];
#pragma unroll
    for (int r = 0; r < 8; ++r) { mrow[r] = -3e38f; lrow[r] = 0.0f; }

    const int nSteps = (qBase + 64) / 32;   // causal: keys 0 .. qBase+63
    for (int s = 0; s < nSteps; ++s) {
        const int kvBase = s * 32;

        // ---- stage K (straight) and V (transposed, permuted keys) ----
        {
            const int key  = tid >> 2;                              // 0..31
            const int pkey = (key < 16) ? (key * 2) : ((key - 16) * 2 + 1);
            const int q4   = (tid & 3) * 32;                        // quarter of head_dim
            const __bf16* ksrc = qkv + (size_t)(kvBase + key) * LDQ + 2048 + h * HD + q4;
#pragma unroll
            for (int j = 0; j < 4; ++j)
                *(uint4*)&Ks[key][q4 + j * 8] = *(const uint4*)(ksrc + j * 8);

            const __bf16* vsrc = qkv + (size_t)(kvBase + key) * LDQ + 4096 + h * HD + q4;
            union { uint4 u[4]; __bf16 hh[32]; } vv;
#pragma unroll
            for (int j = 0; j < 4; ++j) vv.u[j] = *(const uint4*)(vsrc + j * 8);
#pragma unroll
            for (int j = 0; j < 32; ++j) Vt[q4 + j][pkey] = vv.hh[j];
        }
        __syncthreads();

        // ---- S = Q * K^T : two 16x16 score tiles, K-loop over d (4 x 32) ----
        v8f s0 = {}, s1 = {};
#pragma unroll
        for (int kc = 0; kc < 4; ++kc) {
            Frag b0, b1;   // B column n = key index; reduction runs contiguous in Ks rows
            b0.u[0] = *(const uint4*)&Ks[l16][kc * 32 + g * 8];
            b0.u[1] = *(const uint4*)&Ks[l16][kc * 32 + 16 + g * 8];
            b1.u[0] = *(const uint4*)&Ks[16 + l16][kc * 32 + g * 8];
            b1.u[1] = *(const uint4*)&Ks[16 + l16][kc * 32 + 16 + g * 8];
            s0 = wmma_bf16(qf[kc], b0, s0);
            s1 = wmma_bf16(qf[kc], b1, s1);
        }

        // ---- causal mask + online softmax (16-lane row reductions) ----
        const int col0 = kvBase + l16;
        const int col1 = kvBase + 16 + l16;
#pragma unroll
        for (int r = 0; r < 8; ++r) {
            const int row = qRowBase + r + 8 * g;
            float x0 = s0[r] * SCALE;
            float x1 = s1[r] * SCALE;
            if (col0 > row) x0 = -3e38f;
            if (col1 > row) x1 = -3e38f;
            float mx = fmaxf(x0, x1);
#pragma unroll
            for (int m = 1; m < 16; m <<= 1) mx = fmaxf(mx, __shfl_xor(mx, m, 32));
            const float mnew = fmaxf(mrow[r], mx);
            const float p0 = __expf(x0 - mnew);
            const float p1 = __expf(x1 - mnew);
            float sum = p0 + p1;
#pragma unroll
            for (int m = 1; m < 16; m <<= 1) sum += __shfl_xor(sum, m, 32);
            const float corr = __expf(mrow[r] - mnew);
            lrow[r] = lrow[r] * corr + sum;
            mrow[r] = mnew;
#pragma unroll
            for (int nb = 0; nb < 8; ++nb) oacc[nb][r] *= corr;
            // packed store: columns (l16, 16+l16) -> permuted positions (2*l16, 2*l16+1)
            union { __bf16 h[2]; unsigned int u; } pk;
            pk.h[0] = (__bf16)p0;
            pk.h[1] = (__bf16)p1;
            *(unsigned int*)&Ps[wave][r + 8 * g][2 * l16] = pk.u;
        }
        __syncthreads();   // Ps visible (also keeps waves in lockstep)

        // ---- O += P * V : A = P (16x32, permuted keys), B from Vt (same order) ----
        Frag pa;
        pa.u[0] = *(const uint4*)&Ps[wave][l16][g * 8];
        pa.u[1] = *(const uint4*)&Ps[wave][l16][16 + g * 8];
#pragma unroll
        for (int nb = 0; nb < 8; ++nb) {
            Frag vb;
            vb.u[0] = *(const uint4*)&Vt[nb * 16 + l16][g * 8];
            vb.u[1] = *(const uint4*)&Vt[nb * 16 + l16][16 + g * 8];
            oacc[nb] = wmma_bf16(pa, vb, oacc[nb]);
        }
        __syncthreads();   // done with Ks/Vt before next stage overwrites
    }

    // ---- normalize and store (bf16, head-concatenated layout) ----
#pragma unroll
    for (int r = 0; r < 8; ++r) {
        const float inv = 1.0f / lrow[r];
        const int row = qRowBase + r + 8 * g;
#pragma unroll
        for (int nb = 0; nb < 8; ++nb) {
            outb[(size_t)row * DMODEL + h * HD + nb * 16 + l16] =
                (__bf16)(oacc[nb][r] * inv);
        }
    }
}

// ---------------------------------------------------------------------------
// Host launcher
// ---------------------------------------------------------------------------
extern "C" void kernel_launch(void* const* d_in, const int* in_sizes, int n_in,
                              void* d_out, int out_size, void* d_ws, size_t ws_size,
                              hipStream_t stream) {
    (void)in_sizes; (void)n_in; (void)out_size; (void)ws_size;
    constexpr int N = 2048, D = 2048, NQKV = 3 * 2048;   // 6144

    const float* x    = (const float*)d_in[0];   // [2048, 2048]
    const float* Wqkv = (const float*)d_in[1];   // [2048, 6144]
    const float* Wout = (const float*)d_in[2];   // [2048, 2048]

    char* ws = (char*)d_ws;
    __bf16* xb    = (__bf16*)(ws);                                   //  8 MB
    __bf16* wqb   = (__bf16*)(ws + (size_t)8  * 1024 * 1024);        // 24 MB
    __bf16* wob   = (__bf16*)(ws + (size_t)32 * 1024 * 1024);        //  8 MB
    __bf16* qkvb  = (__bf16*)(ws + (size_t)40 * 1024 * 1024);        // 24 MB
    __bf16* attnb = (__bf16*)(ws + (size_t)64 * 1024 * 1024);        //  8 MB

    // 1) fp32 -> bf16 conversions
    {
        int n4;
        n4 = (N * D) / 4;
        cvt_f32_bf16<<<(n4 + 255) / 256, 256, 0, stream>>>(x, xb, n4);
        n4 = (D * NQKV) / 4;
        cvt_f32_bf16<<<(n4 + 255) / 256, 256, 0, stream>>>(Wqkv, wqb, n4);
        n4 = (D * D) / 4;
        cvt_f32_bf16<<<(n4 + 255) / 256, 256, 0, stream>>>(Wout, wob, n4);
    }

    // 2) QKV projection: qkv[2048,6144] = x[2048,2048] @ Wqkv[2048,6144]
    {
        dim3 grid(NQKV / 128, N / 64);
        gemm_bf16_wmma<true><<<grid, 256, 0, stream>>>(xb, wqb, (void*)qkvb, N, NQKV, D);
    }

    // 3) causal flash attention over 16 heads
    {
        dim3 grid(N / 64, 16);
        flash_attn_wmma<<<grid, 128, 0, stream>>>(qkvb, attnb);
    }

    // 4) output projection: out[2048,2048] = attn[2048,2048] @ Wout  (fp32 out)
    {
        dim3 grid(D / 128, N / 64);
        gemm_bf16_wmma<false><<<grid, 256, 0, stream>>>(attnb, wob, d_out, N, D, D);
    }
}
